// Net_33414845563044
// MI455X (gfx1250) — matrix operand order repaired and checked by reading, not compiled, not run on hardware
//
#include <hip/hip_runtime.h>
#include <math.h>

// ---------------- problem constants (match reference) ----------------
#define NN   30000
#define EE   300000
#define GG   256
#define DIN  2050
#define DD   256
#define KP1  2080           // DIN padded to multiple of 32
#define BM   256            // GEMM block rows (8 waves * 32)
#define BN   64             // GEMM block cols (wave col tile)
#define NP   30208          // NN padded to multiple of BM

#if __has_builtin(__builtin_amdgcn_global_load_async_to_lds_b128)
#define HAS_ASYNC_LDS 1
#else
#define HAS_ASYNC_LDS 0
#endif

// ---------------- types ----------------
typedef __attribute__((ext_vector_type(16))) __bf16          v16bf;
typedef __attribute__((ext_vector_type(8)))  float           v8f;
typedef __attribute__((ext_vector_type(8)))  unsigned short  u16x8;
typedef __attribute__((ext_vector_type(4)))  int             v4i;
typedef __attribute__((address_space(1))) v4i*  gv4i_p;   // global 16B vector ptr
typedef __attribute__((address_space(3))) v4i*  lv4i_p;   // LDS 16B vector ptr

union BFrag {
    struct { u16x8 lo, hi; } u;
    v16bf v;
};

static __device__ __forceinline__ unsigned short f32_to_bf16(float f) {
    unsigned int u = __float_as_uint(f);
    u += 0x7FFFu + ((u >> 16) & 1u);   // round-to-nearest-even
    return (unsigned short)(u >> 16);
}

static __device__ __forceinline__ void wait_asynccnt0() {
#if __has_builtin(__builtin_amdgcn_s_wait_asynccnt)
    __builtin_amdgcn_s_wait_asynccnt(0);
#else
    asm volatile("s_wait_asynccnt 0" ::: "memory");
#endif
}

// ---------------- WMMA GEMM: C = A @ B^T(+bias)(+relu) ----------------
// A:  [>=ceil(M/BM)*BM, Kp] row-major bf16 (as u16)  -- pad rows readable
// Bt: [Nc, Kp] row-major bf16 (weights pre-transposed), Kp % 32 == 0, Nc % 64 == 0
// Block: 256 threads = 8 waves; block tile 256x64; wave tile 32x64 (2x4 WMMA tiles).
// B k-slab (64x32 bf16 = 4KB) double-buffered in LDS, streamed with
// GLOBAL_LOAD_ASYNC_TO_LDS_B128 (ASYNCcnt) when available.
__global__ __launch_bounds__(256)
void gemm_bf16_wmma(const unsigned short* __restrict__ A,
                    const unsigned short* __restrict__ Bt,
                    const float* __restrict__ bias,
                    float* __restrict__ Cf, int ldf,
                    unsigned short* __restrict__ Cb, int ldb, int cob,
                    int M, int Nc, int Kp, int relu)
{
    __shared__ unsigned short Bs[2][BN * 32];  // double-buffered [col][k] slab

    int ntn  = Nc / BN;
    int bidm = blockIdx.x / ntn;
    int bidn = blockIdx.x - bidm * ntn;
    int col0 = bidn * BN;

    int t    = threadIdx.x;
    int wave = t >> 5;
    int lane = t & 31;
    int half = lane >> 4;           // 0: lanes 0-15, 1: lanes 16-31
    int r    = lane & 15;

    int wrow = bidm * BM + wave * 32;          // wave's 32-row tile base
    const unsigned short* Arow0 = A + (size_t)(wrow + r) * Kp;
    const unsigned short* Arow1 = A + (size_t)(wrow + 16 + r) * Kp;

    // cooperative B-slab load mapping: thread -> (col, 8 k-values = 16B)
    int bc = t >> 2;                // 0..63 (column within slab)
    int bk = (t & 3) << 3;          // 0/8/16/24
    const unsigned short* Bsrc = Bt + (size_t)(col0 + bc) * Kp + bk;
    int slab = bc * 32 + bk;        // element offset inside one slab

    v8f acc[2][4] = {};
    int nk = Kp >> 5;               // number of 32-wide K steps

#if HAS_ASYNC_LDS
    // prologue: stream slab 0 into buffer 0 (global -> LDS, no VGPR round trip)
    __builtin_amdgcn_global_load_async_to_lds_b128(
        (gv4i_p)(Bsrc), (lv4i_p)&Bs[0][slab], 0, 0);
#endif

    for (int i = 0; i < nk; ++i) {
        int b = i & 1;
        int k0 = i << 5;
#if HAS_ASYNC_LDS
        wait_asynccnt0();            // this wave's portion of slab i landed
        __syncthreads();             // whole slab i landed; buffer b^1 free
        if (i + 1 < nk) {            // stream slab i+1 while computing slab i
            __builtin_amdgcn_global_load_async_to_lds_b128(
                (gv4i_p)(Bsrc + ((i + 1) << 5)), (lv4i_p)&Bs[b ^ 1][slab], 0, 0);
        }
#else
        __syncthreads();             // slab buffer reusable
        *(u16x8*)&Bs[b][slab] = *(const u16x8*)(Bsrc + k0);
        __syncthreads();             // slab ready
#endif

        // A fragments (16-bit A 16x32 layout: lanes<16 hold K 0..7 & 16..23,
        // lanes>=16 hold K 8..15 & 24..31)
        BFrag a0, a1;
        a0.u.lo = *(const u16x8*)(Arow0 + k0 + half * 8);
        a0.u.hi = *(const u16x8*)(Arow0 + k0 + half * 8 + 16);
        a1.u.lo = *(const u16x8*)(Arow1 + k0 + half * 8);
        a1.u.hi = *(const u16x8*)(Arow1 + k0 + half * 8 + 16);
        __builtin_prefetch(Arow0 + k0 + 512, 0, 1);           // global_prefetch_b8
        __builtin_prefetch(Arow1 + k0 + 512, 0, 1);

        #pragma unroll
        for (int j = 0; j < 4; ++j) {
            // B fragment from LDS (16-bit B 32x16: lanes<16 K 0..7 & 16..23, lanes>=16 K 8..15 & 24..31)
            BFrag bb;
            const unsigned short* bp = &Bs[b][(j * 16 + r) * 32 + half * 8];
            bb.u.lo = *(const u16x8*)(bp);
            bb.u.hi = *(const u16x8*)(bp + 16);
            acc[0][j] = __builtin_amdgcn_wmma_f32_16x16x32_bf16(
                            false, a0.v, false, bb.v, (short)0, acc[0][j], false, false);
            acc[1][j] = __builtin_amdgcn_wmma_f32_16x16x32_bf16(
                            false, a1.v, false, bb.v, (short)0, acc[1][j], false, false);
        }
    }

    // epilogue: C/D f32 layout: VGPR i -> row = i + 8*half, col = lane&15
    #pragma unroll
    for (int mt = 0; mt < 2; ++mt) {
        #pragma unroll
        for (int j = 0; j < 4; ++j) {
            #pragma unroll
            for (int i = 0; i < 8; ++i) {
                int rr = wrow + mt * 16 + half * 8 + i;
                int cc = col0 + j * 16 + r;
                float v = acc[mt][j][i];
                if (bias) v += bias[cc];
                if (relu) v = fmaxf(v, 0.0f);
                if (rr < M) {
                    if (Cf) Cf[(size_t)rr * ldf + cc] = v;
                    if (Cb) Cb[(size_t)rr * ldb + cob + cc] = f32_to_bf16(v);
                }
            }
        }
    }
}

// ---------------- conversion / fill kernels ----------------
__global__ void cvt_pad_bf16(const float* __restrict__ x, unsigned short* __restrict__ o,
                             int M, int K, int Kp)
{
    size_t total = (size_t)M * Kp;
    size_t stride = (size_t)gridDim.x * blockDim.x;
    for (size_t i = blockIdx.x * (size_t)blockDim.x + threadIdx.x; i < total; i += stride) {
        size_t m = i / Kp; int k = (int)(i - m * Kp);
        o[i] = (k < K) ? f32_to_bf16(x[m * K + k]) : (unsigned short)0;
    }
}

// W [K,Nc] row-major -> Wt [Nc,Kp] row-major (transposed, zero-padded)
__global__ void cvt_w_bf16(const float* __restrict__ W, unsigned short* __restrict__ Wt,
                           int K, int Nc, int Kp)
{
    size_t total = (size_t)Nc * Kp;
    size_t stride = (size_t)gridDim.x * blockDim.x;
    for (size_t i = blockIdx.x * (size_t)blockDim.x + threadIdx.x; i < total; i += stride) {
        int n = (int)(i / Kp); int k = (int)(i - (size_t)n * Kp);
        Wt[i] = (k < K) ? f32_to_bf16(W[(size_t)k * Nc + n]) : (unsigned short)0;
    }
}

__global__ void fill_f32(float* p, float v, size_t n) {
    size_t stride = (size_t)gridDim.x * blockDim.x;
    for (size_t i = blockIdx.x * (size_t)blockDim.x + threadIdx.x; i < n; i += stride) p[i] = v;
}
__global__ void fill_i32(int* p, int v, size_t n) {
    size_t stride = (size_t)gridDim.x * blockDim.x;
    for (size_t i = blockIdx.x * (size_t)blockDim.x + threadIdx.x; i < n; i += stride) p[i] = v;
}

// ---------------- graph kernels ----------------
__global__ void deg_accum(const int* __restrict__ ei, int E, float* __restrict__ deg) {
    const int* dst = ei + E;
    size_t stride = (size_t)gridDim.x * blockDim.x;
    for (size_t e = blockIdx.x * (size_t)blockDim.x + threadIdx.x; e < (size_t)E; e += stride)
        atomicAdd(&deg[dst[e]], 1.0f);
}
__global__ void make_dinv(float* __restrict__ d, int n) {   // in-place: d = rsqrt(deg+1)
    size_t stride = (size_t)gridDim.x * blockDim.x;
    for (size_t i = blockIdx.x * (size_t)blockDim.x + threadIdx.x; i < (size_t)n; i += stride)
        d[i] = rsqrtf(d[i] + 1.0f);
}
// agg[dst] += h[src] * dinv[src]*dinv[dst]; one thread = (edge, 4 features)
__global__ void edge_agg(const int* __restrict__ ei, int E,
                         const float* __restrict__ h, const float* __restrict__ dinv,
                         float* __restrict__ agg)
{
    size_t total = (size_t)E * 64;
    size_t stride = (size_t)gridDim.x * blockDim.x;
    for (size_t i = blockIdx.x * (size_t)blockDim.x + threadIdx.x; i < total; i += stride) {
        int e = (int)(i >> 6);
        int q = (int)(i & 63) << 2;
        int s = ei[e], tt = ei[E + e];
        float nm = dinv[s] * dinv[tt];
        const float* hs = h + (size_t)s * DD + q;
        float* at = agg + (size_t)tt * DD + q;
        atomicAdd(at + 0, hs[0] * nm);
        atomicAdd(at + 1, hs[1] * nm);
        atomicAdd(at + 2, hs[2] * nm);
        atomicAdd(at + 3, hs[3] * nm);
    }
}
// relu(agg + h*dinv^2 + bias) -> bf16 into concat buffer at column offset
__global__ void gcn_fin(const float* __restrict__ agg, const float* __restrict__ h,
                        const float* __restrict__ dinv, const float* __restrict__ bias,
                        unsigned short* __restrict__ cat, int ldcat, int coff)
{
    size_t total = (size_t)NN * DD;
    size_t stride = (size_t)gridDim.x * blockDim.x;
    for (size_t i = blockIdx.x * (size_t)blockDim.x + threadIdx.x; i < total; i += stride) {
        int d = (int)(i & (DD - 1));
        size_t n = i >> 8;
        float di = dinv[n];
        float v = agg[i] + h[i] * di * di + bias[d];
        v = fmaxf(v, 0.0f);
        cat[n * ldcat + coff + d] = f32_to_bf16(v);
    }
}

// ---------------- pooling ----------------
__global__ void pool_sum(const float* __restrict__ x, const int* __restrict__ idx,
                         float* __restrict__ sums)
{
    size_t total = (size_t)NN * 64;
    size_t stride = (size_t)gridDim.x * blockDim.x;
    for (size_t i = blockIdx.x * (size_t)blockDim.x + threadIdx.x; i < total; i += stride) {
        int n = (int)(i >> 6);
        int q = (int)(i & 63) << 2;
        int g = idx[n];
        const float* xs = x + (size_t)n * DD + q;
        float* sg = sums + (size_t)g * DD + q;
        atomicAdd(sg + 0, xs[0]);
        atomicAdd(sg + 1, xs[1]);
        atomicAdd(sg + 2, xs[2]);
        atomicAdd(sg + 3, xs[3]);
    }
}
__global__ void pool_cnt(const int* __restrict__ idx, float* __restrict__ cnt) {
    size_t stride = (size_t)gridDim.x * blockDim.x;
    for (size_t n = blockIdx.x * (size_t)blockDim.x + threadIdx.x; n < (size_t)NN; n += stride)
        atomicAdd(&cnt[idx[n]], 1.0f);
}
__global__ void pool_out(const float* __restrict__ sums, const float* __restrict__ cnt,
                         unsigned short* __restrict__ cat, int ldcat, int coff)
{
    size_t total = (size_t)GG * DD;
    size_t stride = (size_t)gridDim.x * blockDim.x;
    for (size_t i = blockIdx.x * (size_t)blockDim.x + threadIdx.x; i < total; i += stride) {
        size_t g = i >> 8; int d = (int)(i & (DD - 1));
        float c = fmaxf(cnt[g], 1.0f);
        cat[g * ldcat + coff + d] = f32_to_bf16(sums[i] / c);
    }
}

// ---------------- masks + final concat ----------------
__global__ void flag_set(const int* __restrict__ idx, int* __restrict__ flag) {
    size_t stride = (size_t)gridDim.x * blockDim.x;
    for (size_t n = blockIdx.x * (size_t)blockDim.x + threadIdx.x; n < (size_t)NN; n += stride)
        flag[idx[n]] = 1;
}
__global__ void pack3_sel(const float* __restrict__ xc, const float* __restrict__ xin,
                          const float* __restrict__ xout, const int* __restrict__ fin,
                          const int* __restrict__ fout, unsigned short* __restrict__ cat3)
{
    size_t total = (size_t)GG * 3 * DD;
    size_t stride = (size_t)gridDim.x * blockDim.x;
    for (size_t i = blockIdx.x * (size_t)blockDim.x + threadIdx.x; i < total; i += stride) {
        size_t g = i / (3 * DD); int c = (int)(i - g * (3 * DD));
        float v;
        if (c < DD)            v = xc[g * DD + c];
        else if (c < 2 * DD)   v = fin[g]  ? xin[g * DD + (c - DD)]      : xc[g * DD + (c - DD)];
        else                   v = fout[g] ? xout[g * DD + (c - 2 * DD)] : xc[g * DD + (c - 2 * DD)];
        cat3[i] = f32_to_bf16(v);
    }
}

// ---------------- log_softmax (one wave32 per row, 256 cols) ----------------
__global__ void log_softmax_rows(const float* __restrict__ in, float* __restrict__ out)
{
    int row = blockIdx.x;
    int lane = threadIdx.x;   // 0..31
    const float* r = in + (size_t)row * DD;
    float v[8];
    float m = -INFINITY;
    #pragma unroll
    for (int i = 0; i < 8; ++i) { v[i] = r[lane + i * 32]; m = fmaxf(m, v[i]); }
    #pragma unroll
    for (int o = 16; o > 0; o >>= 1) m = fmaxf(m, __shfl_xor(m, o, 32));
    float s = 0.0f;
    #pragma unroll
    for (int i = 0; i < 8; ++i) s += __expf(v[i] - m);
    #pragma unroll
    for (int o = 16; o > 0; o >>= 1) s += __shfl_xor(s, o, 32);
    float ls = __logf(s);
    #pragma unroll
    for (int i = 0; i < 8; ++i) out[(size_t)row * DD + lane + i * 32] = v[i] - m - ls;
}

// ================= host orchestration =================
static inline int nb256(size_t work) {
    size_t b = (work + 255) / 256;
    if (b > 65535) b = 65535;
    if (b < 1) b = 1;
    return (int)b;
}

extern "C" void kernel_launch(void* const* d_in, const int* in_sizes, int n_in,
                              void* d_out, int out_size, void* d_ws, size_t ws_size,
                              hipStream_t stream)
{
    // ---- inputs ----
    const float* x_orig = (const float*)d_in[0];
    const float* x_out_ = (const float*)d_in[1];
    const float* x_in_  = (const float*)d_in[2];
    const float* w11 = (const float*)d_in[3];  const float* b11 = (const float*)d_in[4];
    const float* w12 = (const float*)d_in[5];  const float* b12 = (const float*)d_in[6];
    const float* w21 = (const float*)d_in[7];  const float* b21 = (const float*)d_in[8];
    const float* w22 = (const float*)d_in[9];  const float* b22 = (const float*)d_in[10];
    const float* m1w1 = (const float*)d_in[11]; const float* m1b1 = (const float*)d_in[12];
    const float* m1w2 = (const float*)d_in[13]; const float* m1b2 = (const float*)d_in[14];
    const float* m2w1 = (const float*)d_in[15]; const float* m2b1 = (const float*)d_in[16];
    const float* m2w2 = (const float*)d_in[17]; const float* m2b2 = (const float*)d_in[18];
    const float* m3w1 = (const float*)d_in[19]; const float* m3b1 = (const float*)d_in[20];
    const float* m3w2 = (const float*)d_in[21]; const float* m3b2 = (const float*)d_in[22];
    const float* mw1  = (const float*)d_in[23]; const float* mb1  = (const float*)d_in[24];
    const float* mw2  = (const float*)d_in[25]; const float* mb2  = (const float*)d_in[26];
    const int* ei1 = (const int*)d_in[27];  const int* ei2 = (const int*)d_in[28];
    const int* eo1 = (const int*)d_in[29];  const int* eo2 = (const int*)d_in[30];
    const int* eii1 = (const int*)d_in[31]; const int* eii2 = (const int*)d_in[32];
    const int* idx1 = (const int*)d_in[33]; const int* idx2 = (const int*)d_in[34];
    const int* idxo1 = (const int*)d_in[35]; const int* idxo2 = (const int*)d_in[36];
    const int* idxi1 = (const int*)d_in[37]; const int* idxi2 = (const int*)d_in[38];
    float* out = (float*)d_out;
    (void)in_sizes; (void)n_in; (void)out_size; (void)ws_size;

    // ---- workspace carve-up ----
    // Buffers consumed as GEMM A-operands get NP (padded) rows; pad rows are
    // never written (contents stay constant across replays) and pad output
    // rows are store-guarded in the GEMM.
    char* ws = (char*)d_ws;
    size_t off = 0;
    auto carve = [&](size_t bytes) -> char* {
        char* p = ws + off;
        off += (bytes + 255) & ~(size_t)255;
        return p;
    };
    unsigned short* XBF   = (unsigned short*)carve((size_t)NP * KP1 * 2);
    float*          H1    = (float*)carve((size_t)NN * DD * 4);
    float*          H2    = (float*)carve((size_t)NN * DD * 4);
    float*          AGG1  = (float*)carve((size_t)NN * DD * 4);
    float*          AGG2  = (float*)carve((size_t)NN * DD * 4);
    unsigned short* CAT   = (unsigned short*)carve((size_t)NP * 2 * DD * 2);
    unsigned short* HID   = (unsigned short*)carve((size_t)NP * DD * 2);
    float*          MID   = (float*)carve((size_t)NN * DD * 4);
    unsigned short* MIDBF = (unsigned short*)carve((size_t)NP * DD * 2);
    float*          DEGA  = (float*)carve((size_t)NN * 4);
    float*          DEGB  = (float*)carve((size_t)NN * 4);
    float*          POOLS = (float*)carve((size_t)GG * DD * 4);
    float*          POOLC = (float*)carve((size_t)GG * 4);
    unsigned short* CATG1 = (unsigned short*)carve((size_t)GG * 2 * DD * 2);
    unsigned short* CATG2 = (unsigned short*)carve((size_t)GG * 2 * DD * 2);
    unsigned short* CATG3 = (unsigned short*)carve((size_t)GG * 2 * DD * 2);
    float*          XC    = (float*)carve((size_t)GG * DD * 4);
    float*          XGIN  = (float*)carve((size_t)GG * DD * 4);
    float*          XGOUT = (float*)carve((size_t)GG * DD * 4);
    unsigned short* M3HID = (unsigned short*)carve((size_t)GG * DD * 2);
    unsigned short* CAT3  = (unsigned short*)carve((size_t)GG * 3 * DD * 2);
    unsigned short* FHID  = (unsigned short*)carve((size_t)GG * 2 * DD * 2);
    float*          LOGIT = (float*)carve((size_t)GG * DD * 4);
    int*            FLIN  = (int*)carve((size_t)GG * 4);
    int*            FLOUT = (int*)carve((size_t)GG * 4);
    unsigned short* W11T  = (unsigned short*)carve((size_t)DD * KP1 * 2);
    unsigned short* W12T  = (unsigned short*)carve((size_t)DD * KP1 * 2);
    unsigned short* W21T  = (unsigned short*)carve((size_t)DD * DD * 2);
    unsigned short* W22T  = (unsigned short*)carve((size_t)DD * DD * 2);
    unsigned short* M1W1T = (unsigned short*)carve((size_t)DD * 2 * DD * 2);
    unsigned short* M1W2T = (unsigned short*)carve((size_t)DD * DD * 2);
    unsigned short* M2W1T = (unsigned short*)carve((size_t)DD * 2 * DD * 2);
    unsigned short* M2W2T = (unsigned short*)carve((size_t)DD * DD * 2);
    unsigned short* M3W1T = (unsigned short*)carve((size_t)DD * 2 * DD * 2);
    unsigned short* M3W2T = (unsigned short*)carve((size_t)DD * DD * 2);
    unsigned short* MW1T  = (unsigned short*)carve((size_t)(2 * DD) * (3 * DD) * 2);
    unsigned short* MW2T  = (unsigned short*)carve((size_t)DD * (2 * DD) * 2);

    // ---- generic GEMM launcher (block tile 256x64) ----
    auto gemm = [&](const unsigned short* A, const unsigned short* Bt, const float* bias,
                    float* Cf, int ldf, unsigned short* Cb, int ldb, int cob,
                    int M, int Nc, int Kp, int relu) {
        int mp = (M + BM - 1) / BM;
        int nblk = mp * (Nc / BN);
        gemm_bf16_wmma<<<nblk, 256, 0, stream>>>(A, Bt, bias, Cf, ldf, Cb, ldb, cob,
                                                 M, Nc, Kp, relu);
    };

    // ---- convert all weights (deterministic per call) ----
    cvt_w_bf16<<<nb256((size_t)DD * KP1), 256, 0, stream>>>(w11, W11T, DIN, DD, KP1);
    cvt_w_bf16<<<nb256((size_t)DD * KP1), 256, 0, stream>>>(w12, W12T, DIN, DD, KP1);
    cvt_w_bf16<<<nb256((size_t)DD * DD), 256, 0, stream>>>(w21, W21T, DD, DD, DD);
    cvt_w_bf16<<<nb256((size_t)DD * DD), 256, 0, stream>>>(w22, W22T, DD, DD, DD);
    cvt_w_bf16<<<nb256((size_t)DD * 2 * DD), 256, 0, stream>>>(m1w1, M1W1T, 2 * DD, DD, 2 * DD);
    cvt_w_bf16<<<nb256((size_t)DD * DD), 256, 0, stream>>>(m1w2, M1W2T, DD, DD, DD);
    cvt_w_bf16<<<nb256((size_t)DD * 2 * DD), 256, 0, stream>>>(m2w1, M2W1T, 2 * DD, DD, 2 * DD);
    cvt_w_bf16<<<nb256((size_t)DD * DD), 256, 0, stream>>>(m2w2, M2W2T, DD, DD, DD);
    cvt_w_bf16<<<nb256((size_t)DD * 2 * DD), 256, 0, stream>>>(m3w1, M3W1T, 2 * DD, DD, 2 * DD);
    cvt_w_bf16<<<nb256((size_t)DD * DD), 256, 0, stream>>>(m3w2, M3W2T, DD, DD, DD);
    cvt_w_bf16<<<nb256((size_t)2 * DD * 3 * DD), 256, 0, stream>>>(mw1, MW1T, 3 * DD, 2 * DD, 3 * DD);
    cvt_w_bf16<<<nb256((size_t)DD * 2 * DD), 256, 0, stream>>>(mw2, MW2T, 2 * DD, DD, 2 * DD);

    const size_t ND = (size_t)NN * DD;

    // ---- one GNN branch ----
    auto run_branch = [&](const float* x, const int* ea, const int* eb,
                          const int* idxA, const int* idxB, unsigned short* catg) {
        // degrees -> dinv (per edge set)
        fill_f32<<<nb256(NN), 256, 0, stream>>>(DEGA, 0.0f, NN);
        fill_f32<<<nb256(NN), 256, 0, stream>>>(DEGB, 0.0f, NN);
        deg_accum<<<nb256(EE), 256, 0, stream>>>(ea, EE, DEGA);
        deg_accum<<<nb256(EE), 256, 0, stream>>>(eb, EE, DEGB);
        make_dinv<<<nb256(NN), 256, 0, stream>>>(DEGA, NN);
        make_dinv<<<nb256(NN), 256, 0, stream>>>(DEGB, NN);

        // ---- layer 1: h = x @ w1{1,2} ----
        cvt_pad_bf16<<<nb256((size_t)NN * KP1), 256, 0, stream>>>(x, XBF, NN, DIN, KP1);
        gemm(XBF, W11T, nullptr, H1, DD, nullptr, 0, 0, NN, DD, KP1, 0);
        gemm(XBF, W12T, nullptr, H2, DD, nullptr, 0, 0, NN, DD, KP1, 0);
        fill_f32<<<nb256(ND), 256, 0, stream>>>(AGG1, 0.0f, ND);
        fill_f32<<<nb256(ND), 256, 0, stream>>>(AGG2, 0.0f, ND);
        edge_agg<<<nb256((size_t)EE * 64), 256, 0, stream>>>(ea, EE, H1, DEGA, AGG1);
        edge_agg<<<nb256((size_t)EE * 64), 256, 0, stream>>>(eb, EE, H2, DEGB, AGG2);
        gcn_fin<<<nb256(ND), 256, 0, stream>>>(AGG1, H1, DEGA, b11, CAT, 2 * DD, 0);
        gcn_fin<<<nb256(ND), 256, 0, stream>>>(AGG2, H2, DEGB, b12, CAT, 2 * DD, DD);
        // MLP1: relu(cat @ m1w1 + b) @ m1w2 + b  -> MID (fp32) + MIDBF (bf16)
        gemm(CAT, M1W1T, m1b1, nullptr, 0, HID, DD, 0, NN, DD, 2 * DD, 1);
        gemm(HID, M1W2T, m1b2, MID, DD, MIDBF, DD, 0, NN, DD, DD, 0);

        // ---- layer 2: h = mid @ w2{1,2} ----
        gemm(MIDBF, W21T, nullptr, H1, DD, nullptr, 0, 0, NN, DD, DD, 0);
        gemm(MIDBF, W22T, nullptr, H2, DD, nullptr, 0, 0, NN, DD, DD, 0);
        fill_f32<<<nb256(ND), 256, 0, stream>>>(AGG1, 0.0f, ND);
        fill_f32<<<nb256(ND), 256, 0, stream>>>(AGG2, 0.0f, ND);
        edge_agg<<<nb256((size_t)EE * 64), 256, 0, stream>>>(ea, EE, H1, DEGA, AGG1);
        edge_agg<<<nb256((size_t)EE * 64), 256, 0, stream>>>(eb, EE, H2, DEGB, AGG2);
        gcn_fin<<<nb256(ND), 256, 0, stream>>>(AGG1, H1, DEGA, b21, CAT, 2 * DD, 0);
        gcn_fin<<<nb256(ND), 256, 0, stream>>>(AGG2, H2, DEGB, b22, CAT, 2 * DD, DD);
        // MLP2 -> branch output MID (fp32)
        gemm(CAT, M2W1T, m2b1, nullptr, 0, HID, DD, 0, NN, DD, 2 * DD, 1);
        gemm(HID, M2W2T, m2b2, MID, DD, nullptr, 0, 0, NN, DD, DD, 0);

        // ---- two segment-mean pools -> bf16 concat [G, 2D] ----
        const int* idxs[2] = { idxA, idxB };
        for (int p = 0; p < 2; ++p) {
            fill_f32<<<nb256((size_t)GG * DD), 256, 0, stream>>>(POOLS, 0.0f, (size_t)GG * DD);
            fill_f32<<<nb256(GG), 256, 0, stream>>>(POOLC, 0.0f, GG);
            pool_sum<<<nb256((size_t)NN * 64), 256, 0, stream>>>(MID, idxs[p], POOLS);
            pool_cnt<<<nb256(NN), 256, 0, stream>>>(idxs[p], POOLC);
            pool_out<<<nb256((size_t)GG * DD), 256, 0, stream>>>(POOLS, POOLC, catg, 2 * DD, p * DD);
        }
    };

    run_branch(x_orig, ei1, ei2, idx1, idx2, CATG1);
    run_branch(x_out_, eo1, eo2, idxo1, idxo2, CATG2);
    run_branch(x_in_, eii1, eii2, idxi1, idxi2, CATG3);

    // ---- m3 MLP on each pooled concat ----
    auto m3 = [&](unsigned short* catg, float* dst) {
        gemm(catg, M3W1T, m3b1, nullptr, 0, M3HID, DD, 0, GG, DD, 2 * DD, 1);
        gemm(M3HID, M3W2T, m3b2, dst, DD, nullptr, 0, 0, GG, DD, DD, 0);
    };
    m3(CATG1, XC);     // x_
    m3(CATG3, XGIN);   // xg_in  (from x_in branch pools)
    m3(CATG2, XGOUT);  // xg_out (from x_out branch pools)

    // ---- presence masks + final concat with selection ----
    fill_i32<<<nb256(GG), 256, 0, stream>>>(FLIN, 0, GG);
    fill_i32<<<nb256(GG), 256, 0, stream>>>(FLOUT, 0, GG);
    flag_set<<<nb256(NN), 256, 0, stream>>>(idxi1, FLIN);
    flag_set<<<nb256(NN), 256, 0, stream>>>(idxo1, FLOUT);
    pack3_sel<<<nb256((size_t)GG * 3 * DD), 256, 0, stream>>>(XC, XGIN, XGOUT, FLIN, FLOUT, CAT3);

    // ---- final MLP + log_softmax ----
    gemm(CAT3, MW1T, mb1, nullptr, 0, FHID, 2 * DD, 0, GG, 2 * DD, 3 * DD, 1);
    gemm(FHID, MW2T, mb2, LOGIT, DD, nullptr, 0, 0, GG, DD, 2 * DD, 0);
    log_softmax_rows<<<GG, 32, 0, stream>>>(LOGIT, out);
}
